// GPR_ATT_32126355374951
// MI455X (gfx1250) — matrix operand order repaired and checked by reading, not compiled, not run on hardware
//
#include <hip/hip_runtime.h>
#include <hip/hip_bf16.h>
#include <math.h>

// Problem constants (match reference)
#define NN 50000
#define NE 800000
#define DIN 512
#define DH 64
#define DOUT 40
#define NL 4

typedef __attribute__((ext_vector_type(16))) __bf16 v16bf;
typedef __attribute__((ext_vector_type(8)))  float  v8f;

// float -> bf16, round-to-nearest-even (storage-only)
__device__ __forceinline__ __bf16 f2bf(float f) {
  unsigned u = __builtin_bit_cast(unsigned, f);
  u += 0x7fffu + ((u >> 16) & 1u);
  unsigned short hs = (unsigned short)(u >> 16);
  return __builtin_bit_cast(__bf16, hs);
}
__device__ __forceinline__ unsigned short f2bf_bits(float f) {
  unsigned u = __builtin_bit_cast(unsigned, f);
  u += 0x7fffu + ((u >> 16) & 1u);
  return (unsigned short)(u >> 16);
}

// ---------------------------------------------------------------------------
// Weight pre-pack: f32 row-major W[K][Ncols] -> bf16 WMMA-B fragment layout
//   packed[((ct*numKB + kb)*32 + lane)*16 + e] = bf16(W[(kb*32 + (e>>3)*16 + (lane>>4)*8 + (e&7))*Ncols + ct*16 + (lane&15)])
// zero-padded where ct*16+(lane&15) >= Ncols (ragged OUT=40). One thread per
// packed element; weights are tiny so this pass is free.
// ---------------------------------------------------------------------------
__global__ void pack_weight_b(const float* __restrict__ W, unsigned short* __restrict__ P,
                              int K, int Ncols, int colTiles) {
  int numKB = K >> 5;
  int total = colTiles * numKB * 32 * 16;
  int idx = blockIdx.x * blockDim.x + threadIdx.x;
  if (idx >= total) return;
  int e    = idx & 15;
  int lane = (idx >> 4) & 31;
  int kb   = (idx >> 9) % numKB;
  int ct   = (idx >> 9) / numKB;
  int l16  = lane & 15;
  int half = lane >> 4;
  int bcol = ct * 16 + l16;
  int kk   = kb * 32 + (e >> 3) * 16 + half * 8 + (e & 7);
  float v = (bcol < Ncols) ? W[(size_t)kk * (size_t)Ncols + bcol] : 0.0f;
  P[idx] = f2bf_bits(v);
}

// ---------------------------------------------------------------------------
// GEMM: C[M,Ncols] = A[M,K] @ B + bias (optional ReLU), B pre-packed bf16.
// bf16 WMMA 16x16x32, f32 accumulate. One wave per 16x16 output tile.
// A loads: 4x contiguous float4 per K-step (b128). B loads: one v16bf
// (32B contiguous per lane, coalesced across the wave). No divergence in
// the K-loop; only the C store is column-masked (OUT=40 case).
// ---------------------------------------------------------------------------
template <int K, int RELU>
__global__ __launch_bounds__(128)
void gemm_bf16_wmma(const float* __restrict__ A, const unsigned short* __restrict__ Bp,
                    const float* __restrict__ bias, float* __restrict__ C,
                    int M, int Ncols, int colTiles) {
  constexpr int numKB = K >> 5;
  const int rowTiles   = M >> 4;
  const int totalTiles = rowTiles * colTiles;
  const int wave = threadIdx.x >> 5;
  const int lane = threadIdx.x & 31;
  const int tile = blockIdx.x * 4 + wave;
  if (tile >= totalTiles) return;               // wave-uniform exit

  const int rowTile = tile / colTiles;
  const int colTile = tile % colTiles;
  const int l16  = lane & 15;
  const int half = lane >> 4;

  const int arow  = rowTile * 16 + l16;
  const int bcol  = colTile * 16 + l16;
  const bool colOK = (bcol < Ncols);

  const float* Ap = A + (size_t)arow * (size_t)K + half * 8;
  const v16bf* Bv = (const v16bf*)Bp + (size_t)colTile * numKB * 32 + lane;

  // accumulator initialized with per-column bias (C layout: VGPR r -> row
  // half*8+r, col = lane&15 -- bias is per-column so broadcast across r)
  float bv = colOK ? bias[bcol] : 0.0f;
  v8f c;
#pragma unroll
  for (int r = 0; r < 8; ++r) c[r] = bv;

#pragma unroll 2
  for (int kb = 0; kb < numKB; ++kb) {
    const int k0 = kb * 32;
    const float4* A4lo = (const float4*)(Ap + k0);        // K = kbA .. kbA+7
    const float4* A4hi = (const float4*)(Ap + k0 + 16);   // K = 16+kbA ..
    float4 a0 = A4lo[0], a1 = A4lo[1];
    float4 a2 = A4hi[0], a3 = A4hi[1];

    v16bf a;
    a[0]  = f2bf(a0.x); a[1]  = f2bf(a0.y); a[2]  = f2bf(a0.z); a[3]  = f2bf(a0.w);
    a[4]  = f2bf(a1.x); a[5]  = f2bf(a1.y); a[6]  = f2bf(a1.z); a[7]  = f2bf(a1.w);
    a[8]  = f2bf(a2.x); a[9]  = f2bf(a2.y); a[10] = f2bf(a2.z); a[11] = f2bf(a2.w);
    a[12] = f2bf(a3.x); a[13] = f2bf(a3.y); a[14] = f2bf(a3.z); a[15] = f2bf(a3.w);

    v16bf b = Bv[kb * 32];

    c = __builtin_amdgcn_wmma_f32_16x16x32_bf16(
            /*neg_a=*/false, a, /*neg_b=*/false, b,
            /*c_mod=*/(short)0, c, /*reuse_a=*/false, /*reuse_b=*/false);
  }

  if (colOK) {
    const int rbase = rowTile * 16 + half * 8;
#pragma unroll
    for (int r = 0; r < 8; ++r) {
      float v = c[r];
      if (RELU) v = fmaxf(v, 0.0f);
      C[(size_t)(rbase + r) * (size_t)Ncols + bcol] = v;
    }
  }
}

// ---------------------------------------------------------------------------
// Elementwise helpers
// ---------------------------------------------------------------------------
__global__ void zero_f32(float* __restrict__ p, int n) {
  int i = blockIdx.x * blockDim.x + threadIdx.x;
  if (i < n) p[i] = 0.0f;
}

// hidden = h * temp[0]; cur = h
__global__ void gpr_init(const float* __restrict__ h, float* __restrict__ hidden,
                         float* __restrict__ cur, const float* __restrict__ temp, int n) {
  int i = blockIdx.x * blockDim.x + threadIdx.x;
  if (i < n) { float v = h[i]; hidden[i] = v * temp[0]; cur[i] = v; }
}

// cur = relu(next); hidden += cur * temp[k]
__global__ void gpr_relu_acc(const float* __restrict__ nxt, float* __restrict__ cur,
                             float* __restrict__ hidden, const float* __restrict__ temp,
                             int k, int n) {
  int i = blockIdx.x * blockDim.x + threadIdx.x;
  if (i < n) {
    float v = fmaxf(nxt[i], 0.0f);
    cur[i] = v;
    hidden[i] += v * temp[k];
  }
}

// ---------------------------------------------------------------------------
// Edge scatter: out[dst[e],f] += m[src[e],f] * w[e]
// thread = (edge, feature); consecutive threads hit consecutive features so
// both the gather read and the f32 global atomic are row-coalesced.
// ---------------------------------------------------------------------------
__global__ __launch_bounds__(256)
void edge_scatter_add(const float* __restrict__ m, const int* __restrict__ src,
                      const int* __restrict__ dst, const float* __restrict__ w,
                      float* __restrict__ out, int E) {
  long long idx = (long long)blockIdx.x * blockDim.x + threadIdx.x;
  if (idx >= (long long)E * DH) return;
  int e = (int)(idx >> 6);
  int f = (int)(idx & 63);
  float val = m[(size_t)src[e] * DH + f] * w[e];
  float* addr = &out[(size_t)dst[e] * DH + f];
  __hip_atomic_fetch_add(addr, val, __ATOMIC_RELAXED, __HIP_MEMORY_SCOPE_AGENT);
}

// ---------------------------------------------------------------------------
// Edge cosine attention fused with edge weight:
//   w2[e] = edge_w[e] * dot(z[src],z[dst]) / (max(|z[src]|,eps)*max(|z[dst]|,eps))
// One wave32 per edge, 2 features per lane, shfl_xor tree reduction.
// ---------------------------------------------------------------------------
__global__ __launch_bounds__(256)
void edge_cosine(const float* __restrict__ z, const int* __restrict__ src,
                 const int* __restrict__ dst, const float* __restrict__ ew,
                 float* __restrict__ w2, int E) {
  int e = blockIdx.x * 8 + (threadIdx.x >> 5);
  int lane = threadIdx.x & 31;
  if (e >= E) return;
  const float* f1 = z + (size_t)src[e] * DH;
  const float* f2 = z + (size_t)dst[e] * DH;
  float a0 = f1[lane], a1 = f1[lane + 32];
  float b0 = f2[lane], b1 = f2[lane + 32];
  float dot = a0 * b0 + a1 * b1;
  float n1  = a0 * a0 + a1 * a1;
  float n2  = b0 * b0 + b1 * b1;
#pragma unroll
  for (int off = 16; off > 0; off >>= 1) {
    dot += __shfl_xor(dot, off, 32);
    n1  += __shfl_xor(n1,  off, 32);
    n2  += __shfl_xor(n2,  off, 32);
  }
  if (lane == 0) {
    float d1 = fmaxf(sqrtf(n1), 1e-8f);
    float d2 = fmaxf(sqrtf(n2), 1e-8f);
    w2[e] = ew[e] * (dot / (d1 * d2));
  }
}

// ---------------------------------------------------------------------------
// Host-side orchestration
// ---------------------------------------------------------------------------
static inline int packed_elems(int K, int colTiles) { return colTiles * K * 16; }

static inline void launch_pack(const float* W, unsigned short* P, int K, int Ncols,
                               hipStream_t s) {
  int colTiles = (Ncols + 15) / 16;
  int total = packed_elems(K, colTiles);
  pack_weight_b<<<(total + 255) / 256, 256, 0, s>>>(W, P, K, Ncols, colTiles);
}

template <int K, int RELU>
static inline void launch_gemm(const float* A, const unsigned short* Bp, const float* bias,
                               float* C, int M, int Ncols, hipStream_t s) {
  int colTiles = (Ncols + 15) / 16;
  int tiles = (M >> 4) * colTiles;
  int blocks = (tiles + 3) / 4;            // 4 waves (tiles) per 128-thread block
  gemm_bf16_wmma<K, RELU><<<blocks, 128, 0, s>>>(A, Bp, bias, C, M, Ncols, colTiles);
}

static void run_gpr(const float* h, const float* w, const int* esrc, const int* edst,
                    const unsigned short* pWs, const float* bs, const float* temp,
                    float* hidden, float* cur, float* m, float* nxt, hipStream_t s) {
  const int NH = NN * DH;
  const int eb = (int)(((long long)NE * DH + 255) / 256);
  const int pstep = packed_elems(DH, DH / 16);           // per-layer packed size
  gpr_init<<<(NH + 255) / 256, 256, 0, s>>>(h, hidden, cur, temp, NH);
  for (int i = 0; i < NL; ++i) {
    launch_gemm<DH, 0>(cur, pWs + (size_t)i * pstep, bs + (size_t)i * DH, m, NN, DH, s);
    zero_f32<<<(NH + 255) / 256, 256, 0, s>>>(nxt, NH);
    edge_scatter_add<<<eb, 256, 0, s>>>(m, esrc, edst, w, nxt, NE);
    gpr_relu_acc<<<(NH + 255) / 256, 256, 0, s>>>(nxt, cur, hidden, temp, i + 1, NH);
  }
}

extern "C" void kernel_launch(void* const* d_in, const int* in_sizes, int n_in,
                              void* d_out, int out_size, void* d_ws, size_t ws_size,
                              hipStream_t stream) {
  const float* x     = (const float*)d_in[0];
  const int*   esrc  = (const int*)  d_in[1];
  const int*   edst  = (const int*)  d_in[2];
  const float* ew    = (const float*)d_in[3];
  const float* W_in  = (const float*)d_in[4];
  const float* b_in  = (const float*)d_in[5];
  const float* Ws    = (const float*)d_in[6];
  const float* bs    = (const float*)d_in[7];
  const float* temp  = (const float*)d_in[8];
  const float* We1   = (const float*)d_in[9];
  const float* be1   = (const float*)d_in[10];
  const float* We2   = (const float*)d_in[11];
  const float* be2   = (const float*)d_in[12];
  const float* W_out = (const float*)d_in[13];
  const float* b_out = (const float*)d_in[14];
  float* out = (float*)d_out;

  const size_t NH = (size_t)NN * DH;
  float* h      = (float*)d_ws;          // [N,H]
  float* hidden = h      + NH;           // [N,H]  gpr accumulator / h_gnn
  float* cur    = hidden + NH;           // [N,H]
  float* m      = cur    + NH;           // [N,H]  gemm scratch / z1
  float* nxt    = m      + NH;           // [N,H]  scatter target / z
  float* w2     = nxt    + NH;           // [E]    attention-modulated weights

  // packed bf16 weight fragments (32B-aligned: all prior regions are
  // multiples of 32 bytes)
  unsigned short* pW_in  = (unsigned short*)(w2 + NE);
  unsigned short* pWs    = pW_in  + packed_elems(DIN, DH / 16);        // 4 layers
  unsigned short* pWe1   = pWs    + (size_t)NL * packed_elems(DH, DH / 16);
  unsigned short* pWe2   = pWe1   + packed_elems(DH, DH / 16);
  unsigned short* pW_out = pWe2   + packed_elems(DH, DH / 16);

  // ---- pre-pack all weights into WMMA B-fragment bf16 layout (tiny) ----
  launch_pack(W_in, pW_in, DIN, DH, stream);
  for (int i = 0; i < NL; ++i)
    launch_pack(Ws + (size_t)i * DH * DH,
                pWs + (size_t)i * packed_elems(DH, DH / 16), DH, DH, stream);
  launch_pack(We1,   pWe1,   DH, DH,   stream);
  launch_pack(We2,   pWe2,   DH, DH,   stream);
  launch_pack(W_out, pW_out, DH, DOUT, stream);

  // h = x @ W_in + b_in                                  [50000,512]@[512,64]
  launch_gemm<DIN, 0>(x, pW_in, b_in, h, NN, DH, stream);

  // ---- GPR pass 1 (plain edge weights) -> hidden = h_gnn ----
  run_gpr(h, ew, esrc, edst, pWs, bs, temp, hidden, cur, m, nxt, stream);

  // z = relu(h_gnn @ We1 + be1) @ We2 + be2   (m = z1, nxt = z)
  launch_gemm<DH, 1>(hidden, pWe1, be1, m,   NN, DH, stream);
  launch_gemm<DH, 0>(m,      pWe2, be2, nxt, NN, DH, stream);

  // w2[e] = ew[e] * cosine(z[src], z[dst])
  edge_cosine<<<(NE + 7) / 8, 256, 0, stream>>>(nxt, esrc, edst, ew, w2, NE);

  // ---- GPR pass 2 (attention-modulated weights) -> hidden = h_gnn2 ----
  run_gpr(h, w2, esrc, edst, pWs, bs, temp, hidden, cur, m, nxt, stream);

  // out = h_gnn2 @ W_out + b_out                         [50000,64]@[64,40]
  launch_gemm<DH, 0>(hidden, pW_out, b_out, out, NN, DOUT, stream);
}
